// SpecializedMultiHeadAttention_31275951850070
// MI455X (gfx1250) — matrix operand order repaired and checked by reading, not compiled, not run on hardware
//
#include <hip/hip_runtime.h>
#include <hip/hip_bf16.h>
#include <math.h>

// ---------------------------------------------------------------------------
// Types for CDNA5 WMMA (gfx1250, wave32)
// ---------------------------------------------------------------------------
typedef __attribute__((ext_vector_type(16))) __bf16 v16bf;
typedef __attribute__((ext_vector_type(8)))  float  v8f;

union BF16x16 {
    v16bf v;
    unsigned short u[16];
    uint4 q[2];
};

__device__ __forceinline__ unsigned short f2bf(float f) {
    unsigned u = __builtin_bit_cast(unsigned, f);
    unsigned r = u + 0x7FFFu + ((u >> 16) & 1u);   // round-to-nearest-even
    return (unsigned short)(r >> 16);
}

// D = A(16x32 bf16) * B(32x16 bf16) + C(16x16 f32)
__device__ __forceinline__ v8f wmma_bf16(const BF16x16& a, const BF16x16& b, v8f c) {
    return __builtin_amdgcn_wmma_f32_16x16x32_bf16(
        /*neg_a=*/false, a.v, /*neg_b=*/false, b.v,
        /*c_mod=*/(short)0, c, /*reuse_a=*/false, /*reuse_b=*/false);
}

// ---------------------------------------------------------------------------
// Problem constants
// ---------------------------------------------------------------------------
#define BATCH   2
#define SEQ     2048
#define DMODEL  1024
#define NHEADS  16
#define NSTD    12
#define DK      64
#define NTOK    (BATCH * SEQ)          // 4096
#define SCALE   0.125f                 // 1/sqrt(64)

// ---------------------------------------------------------------------------
// fp32 -> bf16 elementwise
// ---------------------------------------------------------------------------
__global__ __launch_bounds__(256) void convert_x_kernel(
        const float* __restrict__ src, unsigned short* __restrict__ dst, int n) {
    int i = blockIdx.x * 256 + threadIdx.x;
    if (i < n) dst[i] = f2bf(src[i]);
}

// W[k][n] fp32 (KxN row-major)  ->  Wt[n][k] bf16 (NxK row-major)
__global__ __launch_bounds__(256) void convert_wt_kernel(
        const float* __restrict__ W, unsigned short* __restrict__ Wt, int K, int N) {
    int i = blockIdx.x * 256 + threadIdx.x;
    if (i < N * K) {
        int n = i / K;
        int k = i - n * K;
        Wt[i] = f2bf(W[(size_t)k * N + n]);
    }
}

// ---------------------------------------------------------------------------
// Generic bf16 WMMA GEMM:  C[M,N] = A[M,K] * Bt[N,K]^T + bias[N]
//   4 waves / workgroup; each wave owns a 16x64 strip of a 64x64 block.
//   Branch-free ping-pong pipeline: two fragment register sets, look-ahead
//   load offsets clamped (uniform s_cselect) so the loop has no control flow
//   and no register copies.
//   mode 0: write fp32 row-major C (output projection -> d_out)
//   mode 1: write bf16 scattered to [B,H,S,DK]    (Q, K)
//   mode 2: write bf16 scattered to [B,H,DK,S]    (V transposed)
// ---------------------------------------------------------------------------
__global__ __launch_bounds__(128) void gemm_bf16_kernel(
        const unsigned short* __restrict__ A,
        const unsigned short* __restrict__ Bt,
        const float* __restrict__ bias,
        void* __restrict__ Cout,
        int M, int N, int K, int mode) {
    const int lane = threadIdx.x & 31;
    const int wv   = threadIdx.x >> 5;
    const int row0 = blockIdx.y * 64 + wv * 16;
    const int col0 = blockIdx.x * 64;

    const int lm  = lane & 15;      // M / N sub-index
    const int hi  = lane >> 4;      // half select

    v8f acc[4];
    const v8f vzero = {0.f,0.f,0.f,0.f,0.f,0.f,0.f,0.f};
#pragma unroll
    for (int t = 0; t < 4; ++t) acc[t] = vzero;

    const unsigned short* arow_p = A + (size_t)(row0 + lm) * K + hi * 8;
    const unsigned short* bcol_p = Bt + (size_t)(col0 + lm) * K + hi * 8;

    auto loadA = [&](int kb, BF16x16& a) {
        a.q[0] = *(const uint4*)(arow_p + kb);
        a.q[1] = *(const uint4*)(arow_p + kb + 16);
    };
    auto loadB = [&](int kb, BF16x16* bf) {
#pragma unroll
        for (int nt = 0; nt < 4; ++nt) {
            const unsigned short* bp = bcol_p + (size_t)(nt * 16) * K + kb;
            bf[nt].q[0] = ((const uint4*)bp)[0];
            bf[nt].q[1] = ((const uint4*)(bp + 16))[0];
        }
    };

    BF16x16 aA, aB, bA[4], bB[4];
    loadA(0, aA);
    loadB(0, bA);
    const int kclamp = K - 32;
    for (int i = 0; i < K; i += 64) {
        const int k1 = i + 32;                       // always < K
        loadA(k1, aB);
        loadB(k1, bB);
        __builtin_prefetch(arow_p + k1 + 32, 0, 1);  // global_prefetch stream-ahead
#pragma unroll
        for (int nt = 0; nt < 4; ++nt)
            acc[nt] = wmma_bf16(aA, bA[nt], acc[nt]);
        const int k2 = (i + 64 <= kclamp) ? (i + 64) : kclamp;  // clamped look-ahead
        loadA(k2, aA);
        loadB(k2, bA);
#pragma unroll
        for (int nt = 0; nt < 4; ++nt)
            acc[nt] = wmma_bf16(aB, bB[nt], acc[nt]);
    }

    // C tile layout: VGPR r, lane L -> M = r + 8*(L>>4), N = L&15
#pragma unroll
    for (int nt = 0; nt < 4; ++nt) {
        int gcol = col0 + nt * 16 + lm;
        float bs = bias[gcol];
#pragma unroll
        for (int r = 0; r < 8; ++r) {
            int grow = row0 + r + 8 * hi;
            float val = acc[nt][r] + bs;
            if (mode == 0) {
                ((float*)Cout)[(size_t)grow * N + gcol] = val;
            } else {
                int b  = grow >> 11;          // / 2048
                int s  = grow & 2047;
                int h  = gcol >> 6;           // / 64
                int dk = gcol & 63;
                size_t idx;
                if (mode == 1)
                    idx = (((size_t)(b * NHEADS + h) * SEQ) + s) * DK + dk;
                else
                    idx = (((size_t)(b * NHEADS + h) * DK) + dk) * SEQ + s;
                ((unsigned short*)Cout)[idx] = f2bf(val);
            }
        }
    }
}

// ---------------------------------------------------------------------------
// Attention kernel: one workgroup (4 waves) per (b, h, 16-row q tile).
//   LDS (dynamic): scores 16x2048 f32 (128 KB) + PV partials 4x16x64 f32
//                  (16 KB) + 160 f32 reduction scratch. Fits 320 KB WGP LDS.
// ---------------------------------------------------------------------------
__global__ __launch_bounds__(128) void attn_kernel(
        const unsigned short* __restrict__ Qb,   // [B,H,S,DK] bf16
        const unsigned short* __restrict__ Kb,   // [B,H,S,DK] bf16
        const unsigned short* __restrict__ Vtb,  // [B,H,DK,S] bf16
        float* __restrict__ w_std,               // [B,12,S,S]
        float* __restrict__ w_spec,              // [B,4,S,S]
        unsigned short* __restrict__ attn_out) { // [NTOK, DMODEL] bf16
    extern __shared__ char smem[];
    float* s_sc  = (float*)smem;                    // 16 * 2048
    float* s_pv  = s_sc + 16 * SEQ;                 // 4 * 16 * 64
    float* s_red = s_pv + 4 * 16 * 64;              // [0..127] partials,
                                                    // [128..143] rowmax,
                                                    // [144..159] rowinv
    const int tid  = threadIdx.x;
    const int lane = tid & 31;
    const int wv   = tid >> 5;
    const int q0   = blockIdx.x * 16;
    const int bh   = blockIdx.y;                    // b*16 + h
    const int b    = bh >> 4;
    const int h    = bh & 15;
    const int lm   = lane & 15;
    const int hi   = lane >> 4;

    const v8f vzero = {0.f,0.f,0.f,0.f,0.f,0.f,0.f,0.f};

    // ---- Pass 1: scores = Q * K^T * scale, into LDS -----------------------
    // Hoist the two Q A-frags (Dk=64 -> 2 wmma K-steps of 32)
    BF16x16 aq[2];
    {
        const unsigned short* qp =
            Qb + ((size_t)bh * SEQ + q0 + lm) * DK + hi * 8;
#pragma unroll
        for (int kb = 0; kb < 2; ++kb) {
            aq[kb].q[0] = *(const uint4*)(qp + kb * 32);
            aq[kb].q[1] = *(const uint4*)(qp + kb * 32 + 16);
        }
    }
    const unsigned short* kbase = Kb + (size_t)bh * SEQ * DK + lm * DK + hi * 8;
    auto loadK = [&](int kt, BF16x16* dst) {
        const unsigned short* kp = kbase + (size_t)kt * 16 * DK;
        dst[0].q[0] = ((const uint4*)kp)[0];
        dst[0].q[1] = ((const uint4*)(kp + 16))[0];
        dst[1].q[0] = ((const uint4*)(kp + 32))[0];
        dst[1].q[1] = ((const uint4*)(kp + 48))[0];
    };
    auto scoreTile = [&](int kt, const BF16x16* km) {
        v8f acc = vzero;
        acc = wmma_bf16(aq[0], km[0], acc);
        acc = wmma_bf16(aq[1], km[1], acc);
        int col = kt * 16 + lm;
#pragma unroll
        for (int r = 0; r < 8; ++r)
            s_sc[(r + 8 * hi) * SEQ + col] = acc[r] * SCALE;
    };
    const int kt0 = wv * 32, kt1 = kt0 + 32;
    BF16x16 kmA[2], kmB[2];
    loadK(kt0, kmA);
    for (int kt = kt0; kt < kt1; kt += 2) {
        loadK(kt + 1, kmB);                               // always < kt1
        scoreTile(kt, kmA);
        const int ktn = (kt + 2 < kt1) ? (kt + 2) : (kt1 - 1);  // clamped
        loadK(ktn, kmA);
        scoreTile(kt + 1, kmB);
    }
    __syncthreads();

    // ---- Pass 2a: row max ------------------------------------------------
    const int rrow = tid & 15;
    const int rseg = tid >> 4;                 // 8 segments of 256
    {
        const float* p = s_sc + rrow * SEQ + rseg * 256;
        float m = -3.402823466e38f;
        for (int i = 0; i < 256; ++i) m = fmaxf(m, p[i]);
        s_red[rseg * 16 + rrow] = m;
    }
    __syncthreads();
    if (tid < 16) {
        float m = s_red[tid];
        for (int sgi = 1; sgi < 8; ++sgi) m = fmaxf(m, s_red[sgi * 16 + tid]);
        s_red[128 + tid] = m;
    }
    __syncthreads();

    // ---- Pass 2b: exp + row sum (scores overwritten with exp) ------------
    {
        float mx = s_red[128 + rrow];
        float* p = s_sc + rrow * SEQ + rseg * 256;
        float sum = 0.f;
        for (int i = 0; i < 256; ++i) {
            float e = __expf(p[i] - mx);
            p[i] = e;
            sum += e;
        }
        s_red[rseg * 16 + rrow] = sum;
    }
    __syncthreads();
    if (tid < 16) {
        float s = s_red[tid];
        for (int sgi = 1; sgi < 8; ++sgi) s += s_red[sgi * 16 + tid];
        s_red[144 + tid] = 1.0f / s;
    }
    __syncthreads();

    // ---- Pass 3: coalesced write of normalized weights -------------------
    float* wout;
    if (h < NSTD)
        wout = w_std  + (((size_t)(b * NSTD + h) * SEQ + q0) * SEQ);
    else
        wout = w_spec + (((size_t)(b * (NHEADS - NSTD) + (h - NSTD)) * SEQ + q0) * SEQ);
    for (int row = 0; row < 16; ++row) {
        float inv = s_red[144 + row];
        const float* src = s_sc + row * SEQ;
        float* dst = wout + (size_t)row * SEQ;
        for (int c = tid; c < SEQ; c += 128)
            dst[c] = src[c] * inv;
    }

    // ---- Pass 4: out = softmax * V (K-split across waves, ping-pong) -----
    v8f acc[4];
#pragma unroll
    for (int t = 0; t < 4; ++t) acc[t] = vzero;

    const unsigned short* vbase = Vtb + (size_t)bh * DK * SEQ + lm * SEQ + hi * 8;
    auto loadV = [&](int kb, BF16x16* dst) {
#pragma unroll
        for (int nt = 0; nt < 4; ++nt) {
            const unsigned short* vp = vbase + (size_t)(nt * 16) * SEQ + kb;
            dst[nt].q[0] = ((const uint4*)vp)[0];
            dst[nt].q[1] = ((const uint4*)(vp + 16))[0];
        }
    };
    auto buildA = [&](int kb, BF16x16& a) {
        const float* sp = s_sc + lm * SEQ + kb + hi * 8;
#pragma unroll
        for (int j = 0; j < 8; ++j) a.u[j]     = f2bf(sp[j]);
#pragma unroll
        for (int j = 0; j < 8; ++j) a.u[8 + j] = f2bf(sp[16 + j]);
    };
    const int k0 = wv * (SEQ / 4), kend = k0 + SEQ / 4;
    BF16x16 vbA[4], vbB[4];
    loadV(k0, vbA);
    for (int kb = k0; kb < kend; kb += 64) {
        loadV(kb + 32, vbB);                              // always < kend
        BF16x16 a0;
        buildA(kb, a0);                                   // LDS work overlaps loads
#pragma unroll
        for (int nt = 0; nt < 4; ++nt)
            acc[nt] = wmma_bf16(a0, vbA[nt], acc[nt]);
        const int kn = (kb + 64 < kend) ? (kb + 64) : (kend - 32);  // clamped
        loadV(kn, vbA);
        BF16x16 a1;
        buildA(kb + 32, a1);
#pragma unroll
        for (int nt = 0; nt < 4; ++nt)
            acc[nt] = wmma_bf16(a1, vbB[nt], acc[nt]);
    }
    // dump partials
#pragma unroll
    for (int nt = 0; nt < 4; ++nt) {
#pragma unroll
        for (int r = 0; r < 8; ++r) {
            int row = r + 8 * hi;
            int col = nt * 16 + lm;
            s_pv[(wv * 16 + row) * 64 + col] = acc[nt][r];
        }
    }
    __syncthreads();

    // reduce 4 partials, scale by 1/rowsum, write bf16 attn_out
    for (int i = tid; i < 16 * 64; i += 128) {
        int row = i >> 6;
        int col = i & 63;
        float v = s_pv[row * 64 + col] + s_pv[(16 + row) * 64 + col]
                + s_pv[(32 + row) * 64 + col] + s_pv[(48 + row) * 64 + col];
        v *= s_red[144 + row];
        attn_out[((size_t)(b * SEQ) + q0 + row) * DMODEL + h * DK + col] = f2bf(v);
    }
}

// ---------------------------------------------------------------------------
// Host launcher
// ---------------------------------------------------------------------------
extern "C" void kernel_launch(void* const* d_in, const int* in_sizes, int n_in,
                              void* d_out, int out_size, void* d_ws, size_t ws_size,
                              hipStream_t stream) {
    const float* x  = (const float*)d_in[0];
    // d_in[1] = input_ids (unused by reference math)
    const float* Wq = (const float*)d_in[2];
    const float* bq = (const float*)d_in[3];
    const float* Wk = (const float*)d_in[4];
    const float* bk = (const float*)d_in[5];
    const float* Wv = (const float*)d_in[6];
    const float* bv = (const float*)d_in[7];
    const float* Wo = (const float*)d_in[8];
    const float* bo = (const float*)d_in[9];

    float* out    = (float*)d_out;                           // [2,2048,1024]
    float* w_std  = out + (size_t)NTOK * DMODEL;             // [2,12,2048,2048]
    float* w_spec = w_std + (size_t)BATCH * NSTD * SEQ * SEQ;// [2,4,2048,2048]

    char* ws = (char*)d_ws;
    const size_t MB = 1024u * 1024u;
    unsigned short* xb  = (unsigned short*)(ws);              //  8 MB
    unsigned short* Wqt = (unsigned short*)(ws +  8 * MB);    //  2 MB
    unsigned short* Wkt = (unsigned short*)(ws + 10 * MB);
    unsigned short* Wvt = (unsigned short*)(ws + 12 * MB);
    unsigned short* Wot = (unsigned short*)(ws + 14 * MB);
    unsigned short* Qb  = (unsigned short*)(ws + 16 * MB);    //  8 MB
    unsigned short* Kb  = (unsigned short*)(ws + 24 * MB);    //  8 MB
    unsigned short* Vtb = (unsigned short*)(ws + 32 * MB);    //  8 MB
    unsigned short* AOb = (unsigned short*)(ws + 40 * MB);    //  8 MB

    // 1) fp32 -> bf16 converts (x plain; weights transposed for B-frag loads)
    convert_x_kernel<<<(NTOK * DMODEL) / 256, 256, 0, stream>>>(x, xb, NTOK * DMODEL);
    convert_wt_kernel<<<(DMODEL * DMODEL) / 256, 256, 0, stream>>>(Wq, Wqt, DMODEL, DMODEL);
    convert_wt_kernel<<<(DMODEL * DMODEL) / 256, 256, 0, stream>>>(Wk, Wkt, DMODEL, DMODEL);
    convert_wt_kernel<<<(DMODEL * DMODEL) / 256, 256, 0, stream>>>(Wv, Wvt, DMODEL, DMODEL);
    convert_wt_kernel<<<(DMODEL * DMODEL) / 256, 256, 0, stream>>>(Wo, Wot, DMODEL, DMODEL);

    // 2) Q/K/V projections (WMMA GEMM), scattering to head layouts
    dim3 ggrid(DMODEL / 64, NTOK / 64);
    gemm_bf16_kernel<<<ggrid, 128, 0, stream>>>(xb, Wqt, bq, Qb,  NTOK, DMODEL, DMODEL, 1);
    gemm_bf16_kernel<<<ggrid, 128, 0, stream>>>(xb, Wkt, bk, Kb,  NTOK, DMODEL, DMODEL, 1);
    gemm_bf16_kernel<<<ggrid, 128, 0, stream>>>(xb, Wvt, bv, Vtb, NTOK, DMODEL, DMODEL, 2);

    // 3) attention: full-row softmax per 16-row q tile in 320KB-class LDS
    size_t lds_bytes = (size_t)(16 * SEQ + 4 * 16 * 64 + 160) * sizeof(float); // 148096
    attn_kernel<<<dim3(SEQ / 16, BATCH * NHEADS), 128, lds_bytes, stream>>>(
        Qb, Kb, Vtb, w_std, w_spec, AOb);

    // 4) output projection -> d_out fp32
    gemm_bf16_kernel<<<ggrid, 128, 0, stream>>>(AOb, Wot, bo, out, NTOK, DMODEL, DMODEL, 0);
}
